// GPTLanguageModel_2774548873663
// MI455X (gfx1250) — compile-verified, matrix operand order checked by reading, not run on hardware
//
#include <hip/hip_runtime.h>
#include <math.h>

typedef unsigned short u16;
typedef unsigned int   u32;
typedef __attribute__((ext_vector_type(4)))  unsigned int u32x4;
typedef __attribute__((ext_vector_type(4)))  float        f32x4;
typedef __attribute__((ext_vector_type(8)))  float        v8f;
typedef __attribute__((ext_vector_type(16))) __bf16       v16bf;
typedef __attribute__((vector_size(16)))     int          v4i;   // matches builtin

#define CV   32000
#define CE   384
#define CH   6
#define CD   64
#define CT   256
#define CB   32
#define CL   6
#define CF   1536
#define CBT  (CB * CT)   // 8192
#define CBH  (CB * CH)   // 192

// ---- CDNA5 async global->LDS copy (falls back to sync copy if absent) -----
#if defined(__has_builtin)
#  if __has_builtin(__builtin_amdgcn_global_load_async_to_lds_b128)
#    define HAVE_ASYNC_LDS 1
#  endif
#endif

static __device__ __forceinline__ void cp16_g2l(void* lds_dst, const void* gsrc) {
#ifdef HAVE_ASYNC_LDS
    __builtin_amdgcn_global_load_async_to_lds_b128(
        (__attribute__((address_space(1))) v4i*)gsrc,
        (__attribute__((address_space(3))) v4i*)lds_dst, 0, 0);
#else
    *(u32x4*)lds_dst = *(const u32x4*)gsrc;
#endif
}

static __device__ __forceinline__ void async_fence() {
#ifdef HAVE_ASYNC_LDS
#  if __has_builtin(__builtin_amdgcn_s_wait_asynccnt)
    __builtin_amdgcn_s_wait_asynccnt(0);
#  else
    asm volatile("s_wait_asynccnt 0x0" ::: "memory");
#  endif
#endif
}

static __device__ __forceinline__ u16 f2bf(float f) {
    u32 u = __builtin_bit_cast(u32, f);
    u += 0x7FFFu + ((u >> 16) & 1u);           // round-to-nearest-even
    return (u16)(u >> 16);
}
static __device__ __forceinline__ u32 pack_bf2(float lo, float hi) {
    return (u32)f2bf(lo) | ((u32)f2bf(hi) << 16);
}

// ---------------------------------------------------------------------------
// Embedding: x[b,t,:] = tok_emb[idx[b,t],:] + pos_emb[t,:]
// ---------------------------------------------------------------------------
__global__ void embed_kernel(const int* __restrict__ idx,
                             const float* __restrict__ tok,
                             const float* __restrict__ pos,
                             float* __restrict__ X) {
    int bt = blockIdx.x;
    int t  = bt % CT;
    int tk = idx[bt];
    int c  = threadIdx.x;
    X[(size_t)bt * CE + c] = tok[(size_t)tk * CE + c] + pos[(size_t)t * CE + c];
}

// ---------------------------------------------------------------------------
// LayerNorm fp32 -> bf16 (population variance, eps=1e-5), 128 thr, 3 cols/thr
// ---------------------------------------------------------------------------
__global__ void ln_bf16_kernel(const float* __restrict__ X,
                               const float* __restrict__ g,
                               const float* __restrict__ b,
                               u16* __restrict__ Y) {
    __shared__ float s1[128], s2[128];
    int row = blockIdx.x, tid = threadIdx.x;
    const float* xr = X + (size_t)row * CE;
    float v0 = xr[tid], v1 = xr[tid + 128], v2 = xr[tid + 256];
    s1[tid] = v0 + v1 + v2;
    s2[tid] = v0 * v0 + v1 * v1 + v2 * v2;
    __syncthreads();
    for (int o = 64; o > 0; o >>= 1) {
        if (tid < o) { s1[tid] += s1[tid + o]; s2[tid] += s2[tid + o]; }
        __syncthreads();
    }
    float mean = s1[0] * (1.0f / CE);
    float var  = s2[0] * (1.0f / CE) - mean * mean;
    float rstd = rsqrtf(var + 1e-5f);
    u16* yr = Y + (size_t)row * CE;
    yr[tid]       = f2bf((v0 - mean) * rstd * g[tid]       + b[tid]);
    yr[tid + 128] = f2bf((v1 - mean) * rstd * g[tid + 128] + b[tid + 128]);
    yr[tid + 256] = f2bf((v2 - mean) * rstd * g[tid + 256] + b[tid + 256]);
}

// ---------------------------------------------------------------------------
// Causal softmax: one 256-thread block per (b,h,tq) row, fp32 in -> bf16 out
// ---------------------------------------------------------------------------
__global__ void softmax_causal_kernel(const float* __restrict__ S,
                                      u16* __restrict__ P) {
    __shared__ float red[256];
    int row = blockIdx.x;
    int tq  = row % CT;
    int tid = threadIdx.x;
    const float* sr = S + (size_t)row * CT;
    float v = (tid <= tq) ? sr[tid] : -INFINITY;
    red[tid] = v; __syncthreads();
    for (int o = 128; o > 0; o >>= 1) {
        if (tid < o) red[tid] = fmaxf(red[tid], red[tid + o]);
        __syncthreads();
    }
    float mx = red[0]; __syncthreads();
    float e = (tid <= tq) ? __expf(v - mx) : 0.0f;
    red[tid] = e; __syncthreads();
    for (int o = 128; o > 0; o >>= 1) {
        if (tid < o) red[tid] += red[tid + o];
        __syncthreads();
    }
    float inv = 1.0f / red[0];
    P[(size_t)row * CT + tid] = f2bf(e * inv);
}

// ---------------------------------------------------------------------------
// Generic batched bf16 WMMA GEMM:  C[z] = alpha * A[z] @ B[z] (+bias)(+res)
//   128 threads (4 waves), block tile 128x64, K step 32, double-buffered LDS,
//   async global->LDS staging overlapped with WMMA.  Each wave computes a
//   64x32 patch = 4x2 tiles -> 8 v_wmma_f32_16x16x32_bf16 per K step.
// BMODE: 0 = B fp32 row-major [K,N] (weights; converted to bf16 during stage)
//        1 = B bf16, transposed access B(k,n)=Bp[n*ldb+k]  (Q@K^T)
//        2 = B bf16 row-major [K,N]                        (P@V)
// z offsets: off = (z/zdiv)*s1 + (z%zdiv)*s2  for A, B, C (res shares C's)
// ---------------------------------------------------------------------------
union Frag { u32x4 q[2]; v16bf v; };

template <int BMODE, bool BIAS, bool RES, bool RELU, bool OUTBF16>
__global__ __launch_bounds__(128)
void gemm_wmma_kernel(const u16* __restrict__ A, const void* __restrict__ Bp,
                      void* __restrict__ Cp, const float* __restrict__ bias,
                      const float* __restrict__ resp,
                      int K, int lda, int ldb, int ldc, float alpha,
                      int zdiv, long long sA1, long long sA2,
                      long long sB1, long long sB2,
                      long long sC1, long long sC2) {
    __shared__ alignas(16) u16 lA [2][128][32];   // M-major: lA[p][m][k]
    __shared__ alignas(16) u16 lBt[2][ 64][32];   // N-major: lBt[p][n][k]

    const int tid  = threadIdx.x;
    const int lane = tid & 31;
    const int wave = tid >> 5;
    const int wm   = (wave >> 1) * 64;            // wave patch: 64 rows
    const int wn   = (wave & 1) * 32;             //             32 cols
    const int r    = lane & 15;
    const int hf   = lane >> 4;

    const int z   = blockIdx.z;
    const int zq  = z / zdiv, zr = z % zdiv;
    const long long offA = (long long)zq * sA1 + (long long)zr * sA2;
    const long long offB = (long long)zq * sB1 + (long long)zr * sB2;
    const long long offC = (long long)zq * sC1 + (long long)zr * sC2;

    const u16* Ab = A + offA + (size_t)(blockIdx.y * 128) * lda;

    // ---- tile staging (fixed trip counts -> no exec-mask branching) ----
    auto stageA = [&](int pb, int k0) {
        const u16* As = Ab + k0;
        #pragma unroll
        for (int i = 0; i < 4; i++) {
            int e = tid + i * 128;             // 0..511 : 128 rows x 4 chunks
            int row = e >> 2, seg = e & 3;
            cp16_g2l(&lA[pb][row][seg * 8], As + (size_t)row * lda + seg * 8);
        }
    };
    auto stageB = [&](int pb, int k0) {
        if constexpr (BMODE == 1) {            // bf16, row copy of B^T rows
            const u16* Bs = (const u16*)Bp + offB +
                            (size_t)(blockIdx.x * 64) * ldb + k0;
            #pragma unroll
            for (int i = 0; i < 2; i++) {
                int e = tid + i * 128;         // 0..255 : 64 rows x 4 chunks
                int row = e >> 2, seg = e & 3;
                cp16_g2l(&lBt[pb][row][seg * 8],
                         Bs + (size_t)row * ldb + seg * 8);
            }
        } else if constexpr (BMODE == 0) {     // fp32 [K,N]: convert+transpose
            // two consecutive K rows at a time -> packed dword LDS stores
            const float* Bs = (const float*)Bp + offB +
                              (size_t)k0 * ldb + blockIdx.x * 64;
            #pragma unroll
            for (int i = 0; i < 2; i++) {
                int e  = tid + i * 128;        // 0..255 : 16 kpairs x 16 nquads
                int k  = (e >> 4) * 2;
                int n0 = (e & 15) * 4;
                f32x4 f0 = *(const f32x4*)(Bs + (size_t)k       * ldb + n0);
                f32x4 f1 = *(const f32x4*)(Bs + (size_t)(k + 1) * ldb + n0);
                #pragma unroll
                for (int j = 0; j < 4; j++)
                    *(u32*)&lBt[pb][n0 + j][k] = pack_bf2(f0[j], f1[j]);
            }
        } else {                               // bf16 [K,N]: transpose
            const u16* Bs = (const u16*)Bp + offB +
                            (size_t)k0 * ldb + blockIdx.x * 64;
            #pragma unroll
            for (int i = 0; i < 2; i++) {
                int e = tid + i * 128;         // 0..255
                int k = e >> 3, n0 = (e & 7) * 8;
                u32x4 d = *(const u32x4*)(Bs + (size_t)k * ldb + n0);
                const u16* dh = (const u16*)&d;
                #pragma unroll
                for (int j = 0; j < 8; j++) lBt[pb][n0 + j][k] = dh[j];
            }
        }
    };

    v8f acc[4][2] = {};

    // prologue: fill buffer 0
    stageA(0, 0);
    stageB(0, 0);
    async_fence();
    __syncthreads();

    int p = 0;
    for (int k0 = 0; k0 < K; k0 += 32, p ^= 1) {
        // prefetch next tile into the other buffer (async, overlaps WMMA)
        if (k0 + 32 < K) {
            stageA(p ^ 1, k0 + 32);
            stageB(p ^ 1, k0 + 32);
        }

        // fragments per the 16-bit 16x32 layout: two ds_load_b128 each
        Frag fa[4], fb[2];
        #pragma unroll
        for (int i = 0; i < 4; i++) {
            const u32x4* ra = (const u32x4*)&lA[p][wm + i * 16 + r][0];
            fa[i].q[0] = ra[hf];
            fa[i].q[1] = ra[hf + 2];
        }
        #pragma unroll
        for (int j = 0; j < 2; j++) {
            const u32x4* rb = (const u32x4*)&lBt[p][wn + j * 16 + r][0];
            fb[j].q[0] = rb[hf];
            fb[j].q[1] = rb[hf + 2];
        }
        #pragma unroll
        for (int i = 0; i < 4; i++)
            #pragma unroll
            for (int j = 0; j < 2; j++)
                acc[i][j] = __builtin_amdgcn_wmma_f32_16x16x32_bf16(
                    false, fa[i].v, false, fb[j].v,
                    (short)0, acc[i][j], false, false);

        async_fence();        // next tile's async copies complete
        __syncthreads();      // all waves done reading buf p / writing p^1
    }

    // ---- epilogue: lane(r,hf) owns column n, rows hf*8+t of each 16x16 tile
    #pragma unroll
    for (int i = 0; i < 4; i++) {
        #pragma unroll
        for (int j = 0; j < 2; j++) {
            int n = blockIdx.x * 64 + wn + j * 16 + r;
            #pragma unroll
            for (int t = 0; t < 8; t++) {
                int m = blockIdx.y * 128 + wm + i * 16 + hf * 8 + t;
                size_t ci = (size_t)offC + (size_t)m * ldc + n;
                float c = acc[i][j][t] * alpha;
                if (BIAS) c += bias[n];
                if (RES)  c += resp[ci];
                if (RELU) c = fmaxf(c, 0.0f);
                if (OUTBF16) ((u16*)Cp)[ci] = f2bf(c);
                else         ((float*)Cp)[ci] = c;
            }
        }
    }
}

// ---------------------------------------------------------------------------
extern "C" void kernel_launch(void* const* d_in, const int* in_sizes, int n_in,
                              void* d_out, int out_size, void* d_ws, size_t ws_size,
                              hipStream_t stream) {
    const int*   idx   = (const int*)  d_in[0];
    const float* tok   = (const float*)d_in[1];
    const float* pos   = (const float*)d_in[2];
    const float* Wq    = (const float*)d_in[3];
    const float* Wk    = (const float*)d_in[4];
    const float* Wv    = (const float*)d_in[5];
    const float* Wo    = (const float*)d_in[6];
    const float* bo    = (const float*)d_in[7];
    const float* ln1s  = (const float*)d_in[8];
    const float* ln1b  = (const float*)d_in[9];
    const float* ln2s  = (const float*)d_in[10];
    const float* ln2b  = (const float*)d_in[11];
    const float* W1    = (const float*)d_in[12];
    const float* b1    = (const float*)d_in[13];
    const float* W2    = (const float*)d_in[14];
    const float* b2    = (const float*)d_in[15];
    const float* lnfs  = (const float*)d_in[16];
    const float* lnfb  = (const float*)d_in[17];
    const float* Wh    = (const float*)d_in[18];
    const float* bh    = (const float*)d_in[19];

    char* w = (char*)d_ws;
    auto alloc = [&](size_t bytes) -> void* {
        void* p = (void*)w;
        w += (bytes + 255) & ~(size_t)255;
        return p;
    };
    float* x  = (float*)alloc((size_t)CBT * CE * 4);       // residual stream
    u16*   hb = (u16*)  alloc((size_t)CBT * CE * 2);       // LN output (bf16)
    u16*   qb = (u16*)  alloc((size_t)CBT * CE * 2);
    u16*   kb = (u16*)  alloc((size_t)CBT * CE * 2);
    u16*   vb = (u16*)  alloc((size_t)CBT * CE * 2);
    float* sc = (float*)alloc((size_t)CBH * CT * CT * 4);  // attn scores
    u16*   pb = (u16*)  alloc((size_t)CBH * CT * CT * 2);  // attn probs (bf16)
    u16*   ao = (u16*)  alloc((size_t)CBT * CE * 2);       // attn output (bf16)
    u16*   f1 = (u16*)  alloc((size_t)CBT * CF * 2);       // FFN hidden (bf16)

    embed_kernel<<<CBT, CE, 0, stream>>>(idx, tok, pos, x);

    for (int l = 0; l < CL; ++l) {
        const size_t wEE = (size_t)l * CE * CE;
        const size_t wEF = (size_t)l * CE * CF;

        // LN1
        ln_bf16_kernel<<<CBT, 128, 0, stream>>>(x, ln1s + l * CE, ln1b + l * CE, hb);

        // Q,K,V = h @ W  -> bf16
        {
            dim3 g(CE / 64, CBT / 128, 1);
            gemm_wmma_kernel<0, false, false, false, true><<<g, 128, 0, stream>>>(
                hb, Wq + wEE, qb, nullptr, nullptr,
                CE, CE, CE, CE, 1.0f, 1, 0, 0, 0, 0, 0, 0);
            gemm_wmma_kernel<0, false, false, false, true><<<g, 128, 0, stream>>>(
                hb, Wk + wEE, kb, nullptr, nullptr,
                CE, CE, CE, CE, 1.0f, 1, 0, 0, 0, 0, 0, 0);
            gemm_wmma_kernel<0, false, false, false, true><<<g, 128, 0, stream>>>(
                hb, Wv + wEE, vb, nullptr, nullptr,
                CE, CE, CE, CE, 1.0f, 1, 0, 0, 0, 0, 0, 0);
        }

        // scores[b,h] = (1/8) * Q_bh @ K_bh^T   (z = b*H+h)
        {
            dim3 g(CT / 64, CT / 128, CBH);
            gemm_wmma_kernel<1, false, false, false, false><<<g, 128, 0, stream>>>(
                qb, kb, sc, nullptr, nullptr,
                CD, CE, CE, CT, 0.125f, CH,
                (long long)CT * CE, CD,          // A: b*T*E + h*D
                (long long)CT * CE, CD,          // B: b*T*E + h*D
                (long long)CH * CT * CT, (long long)CT * CT); // C: z*T*T
        }

        // causal softmax -> bf16 probs
        softmax_causal_kernel<<<CBH * CT, 256, 0, stream>>>(sc, pb);

        // attn_out[b,h] = P_bh @ V_bh
        {
            dim3 g(CD / 64, CT / 128, CBH);
            gemm_wmma_kernel<2, false, false, false, true><<<g, 128, 0, stream>>>(
                pb, vb, ao, nullptr, nullptr,
                CT, CT, CE, CE, 1.0f, CH,
                (long long)CH * CT * CT, (long long)CT * CT,  // A: z*T*T
                (long long)CT * CE, CD,                       // B: b*T*E + h*D
                (long long)CT * CE, CD);                      // C: b*T*E + h*D
        }

        // x = x + attn_out @ Wo + bo
        {
            dim3 g(CE / 64, CBT / 128, 1);
            gemm_wmma_kernel<0, true, true, false, false><<<g, 128, 0, stream>>>(
                ao, Wo + wEE, x, bo + l * CE, x,
                CE, CE, CE, CE, 1.0f, 1, 0, 0, 0, 0, 0, 0);
        }

        // LN2
        ln_bf16_kernel<<<CBT, 128, 0, stream>>>(x, ln2s + l * CE, ln2b + l * CE, hb);

        // f1 = relu(h @ W1 + b1) -> bf16
        {
            dim3 g(CF / 64, CBT / 128, 1);
            gemm_wmma_kernel<0, true, false, true, true><<<g, 128, 0, stream>>>(
                hb, W1 + wEF, f1, b1 + (size_t)l * CF, nullptr,
                CE, CE, CF, CF, 1.0f, 1, 0, 0, 0, 0, 0, 0);
        }

        // x = x + f1 @ W2 + b2
        {
            dim3 g(CE / 64, CBT / 128, 1);
            gemm_wmma_kernel<0, true, true, false, false><<<g, 128, 0, stream>>>(
                f1, W2 + wEF, x, b2 + (size_t)l * CE, x,
                CF, CF, CE, CE, 1.0f, 1, 0, 0, 0, 0, 0, 0);
        }
    }

    // final LN + LM head: logits = lnf(x) @ Wh + bh  (fp32 out -> d_out)
    ln_bf16_kernel<<<CBT, 128, 0, stream>>>(x, lnfs, lnfb, hb);
    {
        dim3 g(CV / 64, CBT / 128, 1);
        gemm_wmma_kernel<0, true, false, false, false><<<g, 128, 0, stream>>>(
            hb, Wh, d_out, bh, nullptr,
            CE, CE, CV, CV, 1.0f, 1, 0, 0, 0, 0, 0, 0);
    }
    (void)in_sizes; (void)n_in; (void)out_size; (void)ws_size;
}
